// GCNNet_48395691492010
// MI455X (gfx1250) — matrix operand order repaired and checked by reading, not compiled, not run on hardware
//
#include <hip/hip_runtime.h>
#include <hip/hip_bf16.h>

// ---------------------------------------------------------------------------
// CDNA5 (gfx1250, wave32) GCN pipeline.
// GEMMs: v_wmma_f32_16x16x32_f16, one wave computes a 16 x N strip
// (NT = N/16 accumulators, A fragment reused across NT WMMAs per K-step).
// Activations/weights pre-cast to f16 so fragment loads are pure b128s.
// ---------------------------------------------------------------------------

typedef __attribute__((ext_vector_type(8)))  _Float16 v8h;
typedef __attribute__((ext_vector_type(16))) _Float16 v16h;
typedef __attribute__((ext_vector_type(8)))  float    v8f;

#define BN_EPS 1e-5f

// ---------------------------------------------------------------------------
// f32 -> f16 cast (activations once per layer, weights once per layer)
// ---------------------------------------------------------------------------
__global__ void cast_f16_kernel(const float* __restrict__ in, _Float16* __restrict__ out,
                                size_t count) {
    size_t idx = (size_t)blockIdx.x * blockDim.x + threadIdx.x;
    if (idx < count) out[idx] = (_Float16)in[idx];
}

// ---------------------------------------------------------------------------
// WMMA GEMM:  H[n x N] = A[n x K] * W^T,  A,W in f16 (W row-major N x K),
// K % 32 == 0, N == 16*NT, n % 16 == 0. One wave per 16-row strip.
//
// Fragment layouts per CDNA5 ISA 7.12.2 (wave32):
//   A (16x32): lane r=lane&15 -> row M=r; half=lane>>4;
//       elems 0..7  -> K = half*8 + e        (contiguous: one b128)
//       elems 8..15 -> K = 16 + half*8 + e   (contiguous: one b128)
//   B (32x16): lane r -> col N=r (== row of W); same K mapping as A
//       (two b128s in W's row)
//   C/D (16x16 f32): lane r -> col; vgpr v -> row M = v + half*8
// ---------------------------------------------------------------------------
template <int NT>
__global__ void __launch_bounds__(256)
wmma_gemm_kernel(const _Float16* __restrict__ A, const _Float16* __restrict__ W,
                 float* __restrict__ H, int n, int K, int totalWaves)
{
    const int N = 16 * NT;
    int wave = (blockIdx.x * blockDim.x + threadIdx.x) >> 5;
    if (wave >= totalWaves) return;              // wave-uniform: EXEC stays all-ones
    int lane = threadIdx.x & 31;
    int half = lane >> 4;
    int r    = lane & 15;
    int m0   = wave << 4;

    const _Float16* __restrict__ Arow = A + (size_t)(m0 + r) * K + half * 8;

    v8f acc[NT];
    #pragma unroll
    for (int t = 0; t < NT; ++t) acc[t] = (v8f){};

    for (int k0 = 0; k0 < K; k0 += 32) {
        v8h alo = *(const v8h*)(Arow + k0);
        v8h ahi = *(const v8h*)(Arow + k0 + 16);
        v16h a = __builtin_shufflevector(alo, ahi,
                     0,1,2,3,4,5,6,7,8,9,10,11,12,13,14,15);
        #pragma unroll
        for (int t = 0; t < NT; ++t) {
            const _Float16* Wrow = W + (size_t)(t * 16 + r) * K + k0 + half * 8;
            v8h blo = *(const v8h*)(Wrow);
            v8h bhi = *(const v8h*)(Wrow + 16);
            v16h b = __builtin_shufflevector(blo, bhi,
                         0,1,2,3,4,5,6,7,8,9,10,11,12,13,14,15);
            acc[t] = __builtin_amdgcn_wmma_f32_16x16x32_f16(
                         false, a, false, b, (short)0, acc[t], false, false);
        }
    }

    #pragma unroll
    for (int t = 0; t < NT; ++t) {
        #pragma unroll
        for (int v = 0; v < 8; ++v) {
            int m = m0 + v + half * 8;
            H[(size_t)m * N + t * 16 + r] = acc[t][v];
        }
    }
}

// ---------------------------------------------------------------------------
// Scalar GEMM for layer 0 (K=4 -> N=64): trivial FLOPs, skip WMMA path.
// ---------------------------------------------------------------------------
__global__ void small_gemm_kernel(const float* __restrict__ A, const float* __restrict__ W,
                                  float* __restrict__ H, int n, int K, int N) {
    size_t idx = (size_t)blockIdx.x * blockDim.x + threadIdx.x;
    if (idx >= (size_t)n * N) return;
    int i = (int)(idx / N);
    int j = (int)(idx - (size_t)i * N);
    const float* a = A + (size_t)i * K;
    const float* w = W + (size_t)j * K;
    float acc = 0.f;
    for (int k = 0; k < K; ++k) acc += a[k] * w[k];
    H[idx] = acc;
}

// ---------------------------------------------------------------------------
// gcn_norm setup kernels
// ---------------------------------------------------------------------------
__global__ void deg_init_kernel(float* deg, int n) {
    int i = blockIdx.x * blockDim.x + threadIdx.x;
    if (i < n) deg[i] = 1.0f;                    // self-loop weight
}

__global__ void deg_edge_kernel(const int* __restrict__ dst,
                                const float* __restrict__ w,
                                float* deg, int nE) {
    int e = blockIdx.x * blockDim.x + threadIdx.x;
    if (e < nE) atomicAdd(&deg[dst[e]], w[e]);
}

__global__ void dis_kernel(float* deg_to_dis, float* selfnorm, int n) {
    int i = blockIdx.x * blockDim.x + threadIdx.x;
    if (i >= n) return;
    float d = deg_to_dis[i];
    float r = (d > 0.f) ? rsqrtf(fmaxf(d, 1e-30f)) : 0.f;
    deg_to_dis[i] = r;
    selfnorm[i]   = r * r;
}

__global__ void norm_kernel(const int* __restrict__ src, const int* __restrict__ dst,
                            const float* __restrict__ w, const float* __restrict__ dis,
                            float* __restrict__ normv, int nE) {
    int e = blockIdx.x * blockDim.x + threadIdx.x;
    if (e < nE) normv[e] = dis[src[e]] * w[e] * dis[dst[e]];
}

// ---------------------------------------------------------------------------
// x1 = x[:,4:22] @ ESG_weight(18x4) + ESG_bias + x[:,0:4]
// ---------------------------------------------------------------------------
__global__ void esg_kernel(const float* __restrict__ x, const float* __restrict__ Wesg,
                           const float* __restrict__ besg, float* __restrict__ x1, int n) {
    int i = blockIdx.x * blockDim.x + threadIdx.x;
    if (i >= n) return;
    const float* xr = x + (size_t)i * 22;
    #pragma unroll
    for (int j = 0; j < 4; ++j) {
        float acc = besg[j] + xr[j];
        #pragma unroll
        for (int k = 0; k < 18; ++k) acc += xr[4 + k] * Wesg[k * 4 + j];
        x1[(size_t)i * 4 + j] = acc;
    }
}

// ---------------------------------------------------------------------------
// agg = bias + selfnorm[i] * h[i]   (self-loop term + bias folded into init)
// ---------------------------------------------------------------------------
__global__ void agg_init_kernel(const float* __restrict__ h, const float* __restrict__ bias,
                                const float* __restrict__ selfnorm, float* __restrict__ agg,
                                int n, int F) {
    size_t idx = (size_t)blockIdx.x * blockDim.x + threadIdx.x;
    if (idx >= (size_t)n * F) return;
    int i = (int)(idx / F);
    int f = (int)(idx - (size_t)i * F);
    agg[idx] = bias[f] + selfnorm[i] * h[idx];
}

// ---------------------------------------------------------------------------
// agg[dst] += norm[e] * h[src]  — one thread per (edge, 4-feature chunk)
// ---------------------------------------------------------------------------
__global__ void edge_agg_kernel(const int* __restrict__ src, const int* __restrict__ dst,
                                const float* __restrict__ normv, const float* __restrict__ h,
                                float* __restrict__ agg, int nE, int F) {
    int chunks = F >> 2;
    long long t = (long long)blockIdx.x * blockDim.x + threadIdx.x;
    if (t >= (long long)nE * chunks) return;
    int e = (int)(t / chunks);
    int c = (int)(t - (long long)e * chunks);
    int s = src[e], d = dst[e];
    float w = normv[e];
    const float4 hv = *(const float4*)(h + (size_t)s * F + (c << 2));
    float* ap = agg + (size_t)d * F + (c << 2);
    atomicAdd(ap + 0, w * hv.x);
    atomicAdd(ap + 1, w * hv.y);
    atomicAdd(ap + 2, w * hv.z);
    atomicAdd(ap + 3, w * hv.w);
}

// ---------------------------------------------------------------------------
// Residual bookkeeping: x1 = agg (+ resacc if add_res); resacc += agg if acc_res
// ---------------------------------------------------------------------------
__global__ void residual_kernel(const float* __restrict__ agg, float* __restrict__ resacc,
                                float* __restrict__ x1, int n, int F,
                                int add_res, int acc_res) {
    size_t idx = (size_t)blockIdx.x * blockDim.x + threadIdx.x;
    if (idx >= (size_t)n * F) return;
    float a = agg[idx];
    float out = a;
    if (add_res) out += resacc[idx];
    if (acc_res) resacc[idx] += a;
    x1[idx] = out;
}

__global__ void zero_kernel(float* p, size_t count) {
    size_t idx = (size_t)blockIdx.x * blockDim.x + threadIdx.x;
    if (idx < count) p[idx] = 0.f;
}

// ---------------------------------------------------------------------------
// Column-wise mean/var (population): one block per feature column.
// ---------------------------------------------------------------------------
__global__ void bn_stats_kernel(const float* __restrict__ x, float* __restrict__ stats,
                                int n, int F) {
    int f = blockIdx.x;
    __shared__ float s1[256];
    __shared__ float s2[256];
    float a = 0.f, b = 0.f;
    for (int i = threadIdx.x; i < n; i += blockDim.x) {
        float v = x[(size_t)i * F + f];
        a += v; b += v * v;
    }
    s1[threadIdx.x] = a; s2[threadIdx.x] = b;
    __syncthreads();
    for (int s = blockDim.x >> 1; s > 0; s >>= 1) {
        if ((int)threadIdx.x < s) {
            s1[threadIdx.x] += s1[threadIdx.x + s];
            s2[threadIdx.x] += s2[threadIdx.x + s];
        }
        __syncthreads();
    }
    if (threadIdx.x == 0) {
        float m = s1[0] / (float)n;
        stats[f]       = m;
        stats[128 + f] = s2[0] / (float)n - m * m;
    }
}

// ---------------------------------------------------------------------------
// Fused (optional batchnorm) + softmax_one over feature axis.
// One block per row; blockDim.x == F (32/64/128).
// ---------------------------------------------------------------------------
__global__ void bn_softmax_kernel(float* __restrict__ x, const float* __restrict__ stats,
                                  const float* __restrict__ gamma, const float* __restrict__ beta,
                                  int F, int use_bn) {
    __shared__ float sm[128];
    int row = blockIdx.x;
    int f = threadIdx.x;
    float y = x[(size_t)row * F + f];
    if (use_bn) {
        float m = stats[f], v = stats[128 + f];
        y = (y - m) * rsqrtf(v + BN_EPS) * gamma[f] + beta[f];
    }
    sm[f] = y;
    __syncthreads();
    for (int s = F >> 1; s > 0; s >>= 1) {
        if (f < s) sm[f] = fmaxf(sm[f], sm[f + s]);
        __syncthreads();
    }
    float mx = sm[0];
    __syncthreads();
    float e = __expf(y - mx);
    sm[f] = e;
    __syncthreads();
    for (int s = F >> 1; s > 0; s >>= 1) {
        if (f < s) sm[f] += sm[f + s];
        __syncthreads();
    }
    x[(size_t)row * F + f] = e / (1.0f + sm[0]);
}

// ---------------------------------------------------------------------------
// out[i] = sum_f batchnorm(x1, gf, bef)[i,f] * out_w[f]     (F == 32)
// ---------------------------------------------------------------------------
__global__ void final_out_kernel(const float* __restrict__ x, const float* __restrict__ stats,
                                 const float* __restrict__ gf, const float* __restrict__ bef,
                                 const float* __restrict__ outw, float* __restrict__ out, int n) {
    int i = blockIdx.x * blockDim.x + threadIdx.x;
    if (i >= n) return;
    float acc = 0.f;
    #pragma unroll
    for (int f = 0; f < 32; ++f) {
        float m = stats[f], v = stats[128 + f];
        float y = (x[(size_t)i * 32 + f] - m) * rsqrtf(v + BN_EPS) * gf[f] + bef[f];
        acc += y * outw[f];
    }
    out[i] = acc;
}

// ---------------------------------------------------------------------------
// Host orchestration
// ---------------------------------------------------------------------------
static inline int cdiv(long long a, int b) { return (int)((a + b - 1) / b); }

extern "C" void kernel_launch(void* const* d_in, const int* in_sizes, int n_in,
                              void* d_out, int out_size, void* d_ws, size_t ws_size,
                              hipStream_t stream) {
    const float* x     = (const float*)d_in[0];
    const int*   ei    = (const int*)  d_in[1];
    const float* eattr = (const float*)d_in[2];
    const float* ESGw  = (const float*)d_in[4];
    const float* ESGb  = (const float*)d_in[5];
    const float* W0 = (const float*)d_in[6];   const float* b0 = (const float*)d_in[7];
    const float* W1 = (const float*)d_in[8];   const float* b1 = (const float*)d_in[9];
    const float* W2 = (const float*)d_in[10];  const float* b2 = (const float*)d_in[11];
    const float* Wm = (const float*)d_in[12];  const float* bm = (const float*)d_in[13];
    const float* Wf = (const float*)d_in[14];  const float* bf = (const float*)d_in[15];
    const float* g0 = (const float*)d_in[16];  const float* be0= (const float*)d_in[17];
    const float* g1 = (const float*)d_in[18];  const float* be1= (const float*)d_in[19];
    const float* g2 = (const float*)d_in[20];  const float* be2= (const float*)d_in[21];
    const float* gm = (const float*)d_in[22];  const float* bem= (const float*)d_in[23];
    const float* gf = (const float*)d_in[24];  const float* bef= (const float*)d_in[25];
    const float* outw = (const float*)d_in[26];

    const int n  = in_sizes[0] / 22;     // 50000 (multiple of 16)
    const int nE = in_sizes[1] / 2;      // 640000
    const int NUM_MID = in_sizes[12] / (64 * 64);
    const int* src = ei;
    const int* dst = ei + nE;
    float* out = (float*)d_out;

    // --- workspace carve-up ---
    float* ws = (float*)d_ws;
    float* dis      = ws;                 // n (deg -> rsqrt in place)
    float* selfnorm = dis + 50048;
    float* normv    = selfnorm + 50048;   // nE
    float* x1       = normv + 640000;     // n*128 max
    float* hbuf     = x1 + (size_t)n * 128;
    float* agg      = hbuf + (size_t)n * 128;
    float* resacc   = agg + (size_t)n * 128;
    float* stats    = resacc + (size_t)n * 64;   // 256 floats
    _Float16* x1h   = (_Float16*)(stats + 256);  // n*128 halfs
    _Float16* wh    = x1h + (size_t)n * 128;     // 8192 halfs (max W = 128*64)

    const int T = 256;

    // --- gcn_norm ---
    deg_init_kernel<<<cdiv(n, T), T, 0, stream>>>(dis, n);
    deg_edge_kernel<<<cdiv(nE, T), T, 0, stream>>>(dst, eattr, dis, nE);
    dis_kernel<<<cdiv(n, T), T, 0, stream>>>(dis, selfnorm, n);
    norm_kernel<<<cdiv(nE, T), T, 0, stream>>>(src, dst, eattr, dis, normv, nE);

    // --- input transform ---
    esg_kernel<<<cdiv(n, T), T, 0, stream>>>(x, ESGw, ESGb, x1, n);
    zero_kernel<<<cdiv((long long)n * 64, T), T, 0, stream>>>(resacc, (size_t)n * 64);

    // --- layer table ---
    const int NLAYERS = 3 + NUM_MID;     // 11
    const float *Ws[16], *bs[16], *gs[16], *bes[16];
    int Kd[16], Nd[16];
    Ws[0]=W0; bs[0]=b0; gs[0]=g0; bes[0]=be0; Kd[0]=4;   Nd[0]=64;
    Ws[1]=W1; bs[1]=b1; gs[1]=g1; bes[1]=be1; Kd[1]=64;  Nd[1]=128;
    Ws[2]=W2; bs[2]=b2; gs[2]=g2; bes[2]=be2; Kd[2]=128; Nd[2]=64;
    for (int i = 0; i < NUM_MID; ++i) {
        Ws[3+i]=Wm + (size_t)i*64*64; bs[3+i]=bm + (size_t)i*64;
        gs[3+i]=gm + (size_t)i*64;    bes[3+i]=bem + (size_t)i*64;
        Kd[3+i]=64; Nd[3+i]=64;
    }

    const int mWaves = n / 16;           // 3125 waves, one per 16-row strip
    const int gemmBlocks = cdiv((long long)mWaves * 32, T);

    for (int L = 0; L < NLAYERS; ++L) {
        int K = Kd[L], N = Nd[L];
        if (K % 32 != 0) {
            small_gemm_kernel<<<cdiv((long long)n * N, T), T, 0, stream>>>(
                x1, Ws[L], hbuf, n, K, N);
        } else {
            cast_f16_kernel<<<cdiv((long long)n * K, T), T, 0, stream>>>(
                x1, x1h, (size_t)n * K);
            cast_f16_kernel<<<cdiv((long long)N * K, T), T, 0, stream>>>(
                Ws[L], wh, (size_t)N * K);
            if (N == 64)
                wmma_gemm_kernel<4><<<gemmBlocks, T, 0, stream>>>(x1h, wh, hbuf, n, K, mWaves);
            else if (N == 128)
                wmma_gemm_kernel<8><<<gemmBlocks, T, 0, stream>>>(x1h, wh, hbuf, n, K, mWaves);
            else
                wmma_gemm_kernel<2><<<gemmBlocks, T, 0, stream>>>(x1h, wh, hbuf, n, K, mWaves);
        }
        agg_init_kernel<<<cdiv((long long)n * N, T), T, 0, stream>>>(
            hbuf, bs[L], selfnorm, agg, n, N);
        edge_agg_kernel<<<cdiv((long long)nE * (N >> 2), T), T, 0, stream>>>(
            src, dst, normv, hbuf, agg, nE, N);
        residual_kernel<<<cdiv((long long)n * N, T), T, 0, stream>>>(
            agg, resacc, x1, n, N, (L >= 3) ? 1 : 0, (L >= 2) ? 1 : 0);
        bn_stats_kernel<<<N, 256, 0, stream>>>(x1, stats, n, N);
        bn_softmax_kernel<<<n, N, 0, stream>>>(x1, stats, gs[L], bes[L], N, 1);
    }

    // --- final conv (64 -> 32), softmax_one, batchnorm, output dot ---
    {
        int K = 64, N = 32;
        cast_f16_kernel<<<cdiv((long long)n * K, T), T, 0, stream>>>(x1, x1h, (size_t)n * K);
        cast_f16_kernel<<<cdiv((long long)N * K, T), T, 0, stream>>>(Wf, wh, (size_t)N * K);
        wmma_gemm_kernel<2><<<gemmBlocks, T, 0, stream>>>(x1h, wh, hbuf, n, K, mWaves);
        agg_init_kernel<<<cdiv((long long)n * N, T), T, 0, stream>>>(
            hbuf, bf, selfnorm, agg, n, N);
        edge_agg_kernel<<<cdiv((long long)nE * (N >> 2), T), T, 0, stream>>>(
            src, dst, normv, hbuf, agg, nE, N);
        residual_kernel<<<cdiv((long long)n * N, T), T, 0, stream>>>(
            agg, resacc, x1, n, N, 0, 0);
        bn_softmax_kernel<<<n, N, 0, stream>>>(x1, stats, gf, bef, N, 0);
        bn_stats_kernel<<<N, 256, 0, stream>>>(x1, stats, n, N);
        final_out_kernel<<<cdiv(n, T), T, 0, stream>>>(x1, stats, gf, bef, outw, out, n);
    }
}
